// HiPoNet_21217138442475
// MI455X (gfx1250) — matrix operand order repaired and checked by reading, compiled, not run on hardware
//
#include <hip/hip_runtime.h>
#include <hip/hip_bf16.h>
#include <stdint.h>

// ---------------------------------------------------------------------------
// HiPoNet diffusion-wavelet scattering, fully fused per (batch, weight-set).
// One workgroup (512 threads = 16 wave32) owns one (b,w) problem and keeps all
// state in LDS (~292KB of the 320KB/WGP). All matmuls via v_wmma_f32_16x16x32_bf16.
// ---------------------------------------------------------------------------

typedef __attribute__((ext_vector_type(16))) __bf16 v16bf;
typedef __attribute__((ext_vector_type(8)))  float  v8f;
typedef __attribute__((ext_vector_type(4)))  unsigned int u32x4;

#define NPTS    1024
#define DIM     32
#define SIGMA_  64.0f
#define THRESH_ 0.2f
#define EPS_    1e-8f

// LDS layout (bytes), total 291600 < 320KB/WGP
#define OFF_CUR    0                        // 1024*32 f32  = 131072  diffusion state
#define OFF_XB16   (OFF_CUR  + 131072)      // 1024*32 bf16 =  65536  reweighted cloud (WMMA operand)
#define OFF_PREV   (OFF_XB16 +  65536)      // 1024*32 bf16 =  65536  previous scale snapshot
#define OFF_DG     (OFF_PREV +  65536)      // 1024 f32               squared norms
#define OFF_DEG    (OFF_DG   +   4096)      // 1024 f32               column sums
#define OFF_MASK   (OFF_DEG  +   4096)      // 1024 f32               mask as float
#define OFF_STAGE  (OFF_MASK +   4096)      // 16 waves * 1KB         per-wave 16x32 bf16 W tile
#define OFF_POOL   (OFF_STAGE+  16384)      // 5*32 f32               pooled feature accumulators
#define OFF_A      (OFF_POOL +    640)      // 32 f32                 normalized alphas
#define OFF_CNT    (OFF_A    +    128)      // mask count
#define SMEM_BYTES (OFF_CNT  +     16)

union Frag16 { v16bf v; u32x4 q[2]; };

// A operand 16x32 (this code's k order): lane holds row (lane&15);
// element h -> K = h + (lane>=16)*16 (contiguous, matches B).
__device__ __forceinline__ v16bf load_a16(const char* base, int row, int lane) {
  Frag16 u;
  const int klo = (lane & 16) ? 32 : 0;           // byte offset of K base
  u.q[0] = *(const u32x4*)(base + row * 64 + klo);
  u.q[1] = *(const u32x4*)(base + row * 64 + klo + 16);
  return u.v;
}
// B operand 32x16 (this code's k order): lane holds column (lane&15); half-wave gives contiguous K block:
// element h -> K = h + (lane>=16)*16.
__device__ __forceinline__ v16bf load_b16(const char* base, int row, int lane) {
  Frag16 u;
  const int klo = (lane & 16) ? 32 : 0;
  u.q[0] = *(const u32x4*)(base + row * 64 + klo);
  u.q[1] = *(const u32x4*)(base + row * 64 + klo + 16);
  return u.v;
}

// W element: exp(-D/sigma), threshold, pairwise mask  (TRANS-class VALU -> co-executes with WMMA)
__device__ __forceinline__ float wfun(float g, float dgn, float dgm, float mn, float mm) {
  float D = dgn + dgm - 2.0f * g;
  float wv = __expf(D * (-1.0f / SIGMA_));
  wv = (wv < THRESH_) ? 0.0f : wv;
  return wv * mn * mm;
}

__global__ __launch_bounds__(512)
void hiponet_fused(const float* __restrict__ pc, const uint8_t* __restrict__ maskp,
                   const float* __restrict__ alphas, float* __restrict__ out)
{
  extern __shared__ char smem[];
  float*  curf = (float*)(smem + OFF_CUR);
  __bf16* xb16 = (__bf16*)(smem + OFF_XB16);
  __bf16* prev = (__bf16*)(smem + OFF_PREV);
  float*  dg   = (float*)(smem + OFF_DG);
  float*  deg  = (float*)(smem + OFF_DEG);
  float*  mskf = (float*)(smem + OFF_MASK);
  float*  pool = (float*)(smem + OFF_POOL);
  float*  aSh  = (float*)(smem + OFF_A);
  float*  cntf = (float*)(smem + OFF_CNT);

  const int tid  = threadIdx.x;
  const int lane = tid & 31;
  const int wid  = tid >> 5;               // 0..15 (wave32)
  const int bw   = blockIdx.x;             // 0..31
  const int b    = bw >> 2, w = bw & 3;
  const int hi8  = (lane & 16) ? 8 : 0;    // C/D layout upper half-wave row offset
  const int ml   = lane & 15;

  // ---- P0: init ----------------------------------------------------------
  if (tid < 32)  aSh[tid] = alphas[w * 32 + tid];
  for (int i = tid; i < NPTS; i += 512) {
    mskf[i] = maskp[b * NPTS + i] ? 1.0f : 0.0f;
    deg[i]  = 0.0f;
  }
  if (tid < 160) pool[tid] = 0.0f;
  if (tid == 0)  cntf[0] = 0.0f;
  __syncthreads();

  // ---- P1: normalized alphas, reweighted cloud Xb (f32 state + bf16 operand)
  float nrm2 = 0.0f;
#pragma unroll
  for (int t = 0; t < 32; ++t) nrm2 += aSh[t] * aSh[t];
  const float myA = aSh[tid & 31] * (sqrtf(32.0f) * __frsqrt_rn(nrm2));
  for (int i = tid; i < NPTS * DIM; i += 512) {      // (i&31) constant per thread
    float x = pc[b * NPTS * DIM + i] * myA;
    curf[i] = x;
    xb16[i] = (__bf16)x;
  }
  __syncthreads();

  // ---- P2: squared norms dg, pooled Xb (group 0), mask count -------------
  for (int n = tid; n < NPTS; n += 512) {
    float s = 0.0f;
#pragma unroll
    for (int t = 0; t < 32; ++t) { float x = curf[n * 32 + t]; s += x * x; }
    dg[n] = s;
  }
  {
    const int dd = tid & 31, nb0 = tid >> 5;
    float px = 0.0f, c = 0.0f;
    for (int k = 0; k < 64; ++k) {
      int n = nb0 + 16 * k;
      px += mskf[n] * curf[n * 32 + dd];
      if (dd == 0) c += mskf[n];
    }
    atomicAdd(&pool[dd], px);
    if (dd == 0) atomicAdd(cntf, c);
  }
  __syncthreads();

  // ---- P3: column sums deg[m] via WMMA Gram tiles ------------------------
  for (int T = wid; T < 64 * 64; T += 16) {
    const int n0 = (T >> 6) << 4, m0 = (T & 63) << 4;
    v16bf ga = load_a16((const char*)xb16, n0 + ml, lane);
    v16bf gb = load_b16((const char*)xb16, m0 + ml, lane);
    v8f z = {};
    v8f g = __builtin_amdgcn_wmma_f32_16x16x32_bf16(false, ga, false, gb, (short)0, z, false, false);
    const float dgm = dg[m0 + ml], mm = mskf[m0 + ml];
    float part = 0.0f;
#pragma unroll
    for (int v = 0; v < 8; ++v) {
      int nr = n0 + v + hi8;
      part += wfun(g[v], dg[nr], dgm, mskf[nr], mm);
    }
    atomicAdd(&deg[m0 + ml], part);
  }
  __syncthreads();
  for (int i = tid; i < NPTS; i += 512) deg[i] = fmaxf(deg[i], EPS_);
  __syncthreads();

  // ---- P4: eight in-place applications of W (P^1,P^2,P^4,P^8) ------------
  // next-state lives entirely in registers: 512 lanes x (4x2 tiles x v8f) = 32K floats.
  for (int step = 1; step <= 8; ++step) {
    v8f acc[4][2];
    {
      v8f z = {};
#pragma unroll
      for (int r = 0; r < 4; ++r) { acc[r][0] = z; acc[r][1] = z; }
    }
    for (int ks = 0; ks < 32; ++ks) {
      const int k0 = ks * 32;
      // B fragments of cur (K = m rows, N = d cols), f32 LDS -> bf16 regs
      v16bf bb0, bb1;
      {
        const int mb = k0 + ((lane & 16) ? 16 : 0);
#pragma unroll
        for (int h = 0; h < 16; ++h) {
          bb0[h] = (__bf16)curf[(mb + h) * 32 + ml];
          bb1[h] = (__bf16)curf[(mb + h) * 32 + 16 + ml];
        }
      }
      // Gram operands + per-column constants for this 32-wide K block
      const int mr0 = k0 + ml, mr1 = mr0 + 16;
      v16bf gb0 = load_b16((const char*)xb16, mr0, lane);
      v16bf gb1 = load_b16((const char*)xb16, mr1, lane);
      const float dgm0 = dg[mr0],  dgm1 = dg[mr1];
      const float mm0  = mskf[mr0], mm1 = mskf[mr1];
      const float id0  = 1.0f / deg[mr0], id1 = 1.0f / deg[mr1];
      char* st = smem + OFF_STAGE + wid * 1024;
#pragma unroll
      for (int r = 0; r < 4; ++r) {
        const int n0 = (wid * 4 + r) * 16;
        v16bf ga = load_a16((const char*)xb16, n0 + ml, lane);
        v8f z = {};
        v8f g0 = __builtin_amdgcn_wmma_f32_16x16x32_bf16(false, ga, false, gb0, (short)0, z, false, false);
        v8f g1 = __builtin_amdgcn_wmma_f32_16x16x32_bf16(false, ga, false, gb1, (short)0, z, false, false);
        // build normalized W tile (16 rows x 32 K) in per-wave LDS, bf16
#pragma unroll
        for (int v = 0; v < 8; ++v) {
          const int nr = n0 + v + hi8;
          float w0 = wfun(g0[v], dg[nr], dgm0, mskf[nr], mm0) * id0;
          float w1 = wfun(g1[v], dg[nr], dgm1, mskf[nr], mm1) * id1;
          if (nr == k0 + ml)      w0 += 0.5f;   // + 0.5*I after normalization
          if (nr == k0 + 16 + ml) w1 += 0.5f;
          *(__bf16*)(st + (v + hi8) * 64 + ml * 2)        = (__bf16)w0;
          *(__bf16*)(st + (v + hi8) * 64 + (16 + ml) * 2) = (__bf16)w1;
        }
        v16bf aw = load_a16(st, ml, lane);      // same-wave DS ops are in-order
        acc[r][0] = __builtin_amdgcn_wmma_f32_16x16x32_bf16(false, aw, false, bb0, (short)0, acc[r][0], false, false);
        acc[r][1] = __builtin_amdgcn_wmma_f32_16x16x32_bf16(false, aw, false, bb1, (short)0, acc[r][1], false, false);
      }
    }
    __syncthreads();                      // all reads of cur complete
#pragma unroll
    for (int r = 0; r < 4; ++r) {
      const int n0 = (wid * 4 + r) * 16;
#pragma unroll
      for (int c = 0; c < 2; ++c)
#pragma unroll
        for (int v = 0; v < 8; ++v)
          curf[(n0 + v + hi8) * 32 + c * 16 + ml] = acc[r][c][v];
    }
    __syncthreads();                      // cur = P^step X visible to all

    // checkpoints at P^1, P^2, P^4, P^8: fold wavelet/scale pools, snapshot prev
    const int j = (step == 1) ? 0 : (step == 2) ? 1 : (step == 4) ? 2 : (step == 8) ? 3 : -1;
    if (j >= 0) {
      const int dd = tid & 31, nb0 = tid >> 5;
      float wsum = 0.0f, ssum = 0.0f;
      for (int k = 0; k < 64; ++k) {
        int n = nb0 + 16 * k;
        float c = curf[n * 32 + dd], mk = mskf[n];
        if (j > 0)  wsum += mk * fabsf((float)prev[n * 32 + dd] - c);
        if (j == 3) ssum += mk * c;
        prev[n * 32 + dd] = (__bf16)c;    // each (n,dd) owned by exactly one thread
      }
      if (j > 0)  atomicAdd(&pool[(2 + (j - 1)) * 32 + dd], wsum);
      if (j == 3) atomicAdd(&pool[1 * 32 + dd], ssum);
    }
    __syncthreads();
  }

  // ---- P5: masked means -> out[b, w, group, dd] --------------------------
  if (tid < 160) {
    const float inv = 1.0f / fmaxf(cntf[0], EPS_);
    out[bw * 160 + tid] = pool[tid] * inv;
  }
}

extern "C" void kernel_launch(void* const* d_in, const int* in_sizes, int n_in,
                              void* d_out, int out_size, void* d_ws, size_t ws_size,
                              hipStream_t stream) {
  (void)in_sizes; (void)n_in; (void)d_ws; (void)ws_size; (void)out_size;
  const float*   pc     = (const float*)d_in[0];    // (8,1024,32) f32
  const uint8_t* maskp  = (const uint8_t*)d_in[1];  // (8,1024) bool
  const float*   alphas = (const float*)d_in[2];    // (4,32) f32
  float*         outp   = (float*)d_out;            // (8, 4*5*32) f32
  hiponet_fused<<<dim3(32), dim3(512), SMEM_BYTES, stream>>>(pc, maskp, alphas, outp);
}
